// CausalMultiHeadSelfAttention_19774029431222
// MI455X (gfx1250) — matrix operand order repaired and checked by reading, not compiled, not run on hardware
//
#include <hip/hip_runtime.h>
#include <hip/hip_bf16.h>
#include <stdint.h>

// ---------------------------------------------------------------------------
// CDNA5 (gfx1250) causal single-head attention, bf16 WMMA pipeline.
// Every matmul is D = A(M,K) x B(N,K)^T via v_wmma_f32_16x16x32_bf16.
// 64x64 wave tile (16 wmma / 16 b128-loads per K-step), 128x256 WG macro tile.
// Loads are SADDR-form: uniform SGPR base + fixed per-lane VGPR offset +
// compile-time K advance folded into the 24-bit instruction ioffset
// (K is a template constant and the K loop is fully unrolled).
// ---------------------------------------------------------------------------

typedef __attribute__((ext_vector_type(16))) __bf16 v16bf;
typedef __attribute__((ext_vector_type(8)))  float  v8f;

#define NEG_INF (-1e30f)

__device__ __forceinline__ uint16_t f32_to_bf16_bits(float f) {
    union { float f; uint32_t u; } x; x.f = f;
    uint32_t r = (x.u + 0x7FFFu + ((x.u >> 16) & 1u)) >> 16;  // RNE
    return (uint16_t)r;
}

union FragU { v16bf v; uint4 q[2]; };

__device__ __forceinline__ uint4 ld128(const char* __restrict__ base, uint32_t off) {
    return *(const uint4*)(base + off);
}

// OUT_MODE: 0 = f32 row-major, 1 = bf16 row-major, 2 = bf16 transposed (ldc = global M)
// CAUSAL: scale then mask col>row with NEG_INF (OUT_MODE 0 only).
template <int OUT_MODE, bool CAUSAL, int K>
__global__ __launch_bounds__(256)
void gemm_wmma_bf16(const uint16_t* __restrict__ A, const uint16_t* __restrict__ B,
                    void* __restrict__ Cv,
                    int lda, int ldb, int ldc,
                    long long sA, long long sB, long long sC, float scale) {
    const int z = blockIdx.z;
    const char* __restrict__ Ab = (const char*)(A + (size_t)z * (size_t)sA);
    const char* __restrict__ Bb = (const char*)(B + (size_t)z * (size_t)sB);

    const int lane = threadIdx.x & 31;
    const int wid  = threadIdx.x >> 5;
    const int wm   = wid & 1;           // 2 waves along M
    const int wn   = wid >> 1;          // 4 waves along N
    const int m0   = blockIdx.y * 128 + wm * 64;
    const int n0   = blockIdx.x * 256 + wn * 64;
    const int half = lane >> 4, r = lane & 15;

    if (CAUSAL && (n0 > m0 + 63)) {
        // whole 64x64 wave tile above the diagonal: write mask, skip math
        float* Cf = (float*)Cv + (size_t)z * (size_t)sC;
#pragma unroll
        for (int mi = 0; mi < 4; ++mi)
#pragma unroll
            for (int ni = 0; ni < 4; ++ni)
#pragma unroll
                for (int i = 0; i < 8; ++i) {
                    int row = m0 + mi * 16 + i + half * 8;
                    int col = n0 + ni * 16 + r;
                    Cf[(size_t)row * ldc + col] = NEG_INF;
                }
        return;
    }

    // Per-lane byte offsets (fixed for the whole K sweep).
    // A 16x32 frag: lane -> row l%16; bytes [aoff+koff, +16) and [aoff+koff+32, +16)
    // B 32x16 frag: lane -> col l%16; bytes [boff+koff, +16) and [boff+koff+32, +16)
    uint32_t aoff[4], boff[4];
#pragma unroll
    for (int i = 0; i < 4; ++i)
        aoff[i] = (uint32_t)(((m0 + 16 * i + r) * lda + half * 8) * 2);
#pragma unroll
    for (int j = 0; j < 4; ++j)
        boff[j] = (uint32_t)(((n0 + 16 * j + r) * ldb + half * 8) * 2);

    v8f c[4][4] = {};
#pragma unroll
    for (int k0 = 0; k0 < K; k0 += 32) {
        const uint32_t kb = (uint32_t)(k0 * 2);   // compile-time under full unroll
        FragU a[4], b[4];
#pragma unroll
        for (int i = 0; i < 4; ++i) {
            a[i].q[0] = ld128(Ab, aoff[i] + kb);
            a[i].q[1] = ld128(Ab, aoff[i] + kb + 32);
        }
#pragma unroll
        for (int j = 0; j < 4; ++j) {
            b[j].q[0] = ld128(Bb, boff[j] + kb);
            b[j].q[1] = ld128(Bb, boff[j] + kb + 32);
        }
        if (((k0 & 63) == 0) && (k0 + 64 < K)) {  // one stream hint per 128B
            __builtin_prefetch(Ab + aoff[0] + kb + 128, 0, 1);
            __builtin_prefetch(Bb + boff[0] + kb + 128, 0, 1);
        }
#pragma unroll
        for (int i = 0; i < 4; ++i)
#pragma unroll
            for (int j = 0; j < 4; ++j)
                c[i][j] = __builtin_amdgcn_wmma_f32_16x16x32_bf16(
                    false, a[i].v, false, b[j].v, (short)0, c[i][j], false, false);
    }

    // C/D frag layout: VGPR i -> row m0+16*mi+i+8*half, col n0+16*ni+r
    if (OUT_MODE == 0) {
        float* Cf = (float*)Cv + (size_t)z * (size_t)sC;
#pragma unroll
        for (int mi = 0; mi < 4; ++mi)
#pragma unroll
            for (int ni = 0; ni < 4; ++ni)
#pragma unroll
                for (int i = 0; i < 8; ++i) {
                    int row = m0 + mi * 16 + i + half * 8;
                    int col = n0 + ni * 16 + r;
                    float v = c[mi][ni][i] * scale;
                    if (CAUSAL && col > row) v = NEG_INF;
                    Cf[(size_t)row * ldc + col] = v;
                }
    } else if (OUT_MODE == 1) {
        uint16_t* Ch = (uint16_t*)Cv + (size_t)z * (size_t)sC;
#pragma unroll
        for (int mi = 0; mi < 4; ++mi)
#pragma unroll
            for (int ni = 0; ni < 4; ++ni)
#pragma unroll
                for (int i = 0; i < 8; ++i) {
                    int row = m0 + mi * 16 + i + half * 8;
                    int col = n0 + ni * 16 + r;
                    Ch[(size_t)row * ldc + col] = f32_to_bf16_bits(c[mi][ni][i] * scale);
                }
    } else {  // transposed bf16 store: Ct[col*ldc + row], per-lane 16B packs
        uint16_t* Ct = (uint16_t*)Cv + (size_t)z * (size_t)sC;
#pragma unroll
        for (int mi = 0; mi < 4; ++mi)
#pragma unroll
            for (int ni = 0; ni < 4; ++ni) {
                int base = m0 + mi * 16 + half * 8;  // 8 consecutive rows
                int col  = n0 + ni * 16 + r;
                union { uint16_t h[8]; uint4 q; } pk;
#pragma unroll
                for (int i = 0; i < 8; ++i) pk.h[i] = f32_to_bf16_bits(c[mi][ni][i] * scale);
                *(uint4*)(Ct + (size_t)col * ldc + base) = pk.q;
            }
    }
}

// ---------------------------------------------------------------------------
// Row softmax over 1024 f32 scores -> bf16 probabilities. One WG (8 waves) per row.
// ---------------------------------------------------------------------------
__global__ __launch_bounds__(256)
void softmax_rows(const float* __restrict__ S, uint16_t* __restrict__ P, int n) {
    const size_t row = blockIdx.x;
    const float* p = S + row * (size_t)n;
    const int t = threadIdx.x;
    const int wid = t >> 5, lane = t & 31;

    float4 v = *(const float4*)(p + t * 4);

    float m = fmaxf(fmaxf(v.x, v.y), fmaxf(v.z, v.w));
#pragma unroll
    for (int off = 16; off > 0; off >>= 1) m = fmaxf(m, __shfl_xor(m, off, 32));

    __shared__ float red[8];
    if (lane == 0) red[wid] = m;
    __syncthreads();
    float rmax = red[0];
#pragma unroll
    for (int i = 1; i < 8; ++i) rmax = fmaxf(rmax, red[i]);
    __syncthreads();

    float e0 = __expf(v.x - rmax), e1 = __expf(v.y - rmax);
    float e2 = __expf(v.z - rmax), e3 = __expf(v.w - rmax);
    float s = e0 + e1 + e2 + e3;
#pragma unroll
    for (int off = 16; off > 0; off >>= 1) s += __shfl_xor(s, off, 32);
    if (lane == 0) red[wid] = s;
    __syncthreads();
    float tot = 0.f;
#pragma unroll
    for (int i = 0; i < 8; ++i) tot += red[i];
    float inv = __frcp_rn(tot);

    union { uint16_t h[4]; uint2 q; } pk;
    pk.h[0] = f32_to_bf16_bits(e0 * inv);
    pk.h[1] = f32_to_bf16_bits(e1 * inv);
    pk.h[2] = f32_to_bf16_bits(e2 * inv);
    pk.h[3] = f32_to_bf16_bits(e3 * inv);
    *(uint2*)(P + row * (size_t)n + t * 4) = pk.q;
}

// ---------------------------------------------------------------------------
// fp32 -> bf16 conversion
// ---------------------------------------------------------------------------
__global__ __launch_bounds__(256)
void cvt_f32_bf16(const float* __restrict__ in, uint16_t* __restrict__ out, size_t n) {
    size_t i = (size_t)blockIdx.x * blockDim.x + threadIdx.x;
    if (i < n) out[i] = f32_to_bf16_bits(in[i]);
}

// ---------------------------------------------------------------------------
extern "C" void kernel_launch(void* const* d_in, const int* in_sizes, int n_in,
                              void* d_out, int out_size, void* d_ws, size_t ws_size,
                              hipStream_t stream) {
    (void)in_sizes; (void)n_in; (void)out_size; (void)ws_size;
    constexpr int B = 8, S = 1024, D = 1024;
    constexpr int BS = B * S;                       // 8192
    constexpr long long SD = (long long)S * D;      // 1,048,576
    constexpr long long SS = (long long)S * S;      // 1,048,576

    const float* x  = (const float*)d_in[0];
    const float* wq = (const float*)d_in[1];
    const float* wk = (const float*)d_in[2];
    const float* wv = (const float*)d_in[3];
    const float* wo = (const float*)d_in[4];
    float* out = (float*)d_out;

    char* w = (char*)d_ws;
    size_t off = 0;
    auto alloc = [&](size_t bytes) -> void* {
        void* p = w + off;
        off += (bytes + 255) & ~(size_t)255;
        return p;
    };
    uint16_t* Xb  = (uint16_t*)alloc((size_t)BS * D * 2);    // 16 MB
    uint16_t* Wqb = (uint16_t*)alloc((size_t)D * D * 2);     //  2 MB
    uint16_t* Wkb = (uint16_t*)alloc((size_t)D * D * 2);
    uint16_t* Wvb = (uint16_t*)alloc((size_t)D * D * 2);
    uint16_t* Wob = (uint16_t*)alloc((size_t)D * D * 2);
    uint16_t* Qb  = (uint16_t*)alloc((size_t)BS * D * 2);    // 16 MB
    uint16_t* Kb  = (uint16_t*)alloc((size_t)BS * D * 2);    // 16 MB
    uint16_t* VT  = (uint16_t*)alloc((size_t)D * BS * 2);    // 16 MB, (D, B*S)
    float*    Sc  = (float*)   alloc((size_t)B * S * S * 4); // 32 MB
    uint16_t* P   = (uint16_t*)alloc((size_t)B * S * S * 2); // 16 MB
    uint16_t* AO  = (uint16_t*)alloc((size_t)BS * D * 2);    // 16 MB

    // 1) convert inputs to bf16
    cvt_f32_bf16<<<(unsigned)(((size_t)BS * D + 255) / 256), 256, 0, stream>>>(x,  Xb,  (size_t)BS * D);
    cvt_f32_bf16<<<(unsigned)(((size_t)D  * D + 255) / 256), 256, 0, stream>>>(wq, Wqb, (size_t)D * D);
    cvt_f32_bf16<<<(unsigned)(((size_t)D  * D + 255) / 256), 256, 0, stream>>>(wk, Wkb, (size_t)D * D);
    cvt_f32_bf16<<<(unsigned)(((size_t)D  * D + 255) / 256), 256, 0, stream>>>(wv, Wvb, (size_t)D * D);
    cvt_f32_bf16<<<(unsigned)(((size_t)D  * D + 255) / 256), 256, 0, stream>>>(wo, Wob, (size_t)D * D);

    // 2) Q = X Wq^T, K = X Wk^T  (bf16, row-major (B*S, D))
    dim3 gProj(D / 256, BS / 128, 1);   // (4, 64)
    gemm_wmma_bf16<1, false, 1024><<<gProj, 256, 0, stream>>>(Xb, Wqb, Qb, D, D, D, 0, 0, 0, 1.0f);
    gemm_wmma_bf16<1, false, 1024><<<gProj, 256, 0, stream>>>(Xb, Wkb, Kb, D, D, D, 0, 0, 0, 1.0f);

    // 3) V^T = (X Wv^T)^T stored as (D, B*S) bf16 so PV is also (M,K)x(N,K)
    gemm_wmma_bf16<2, false, 1024><<<gProj, 256, 0, stream>>>(Xb, Wvb, VT, D, D, /*ldc=*/BS, 0, 0, 0, 1.0f);

    // 4) scores = Q K^T / 32, causal-masked, f32, batched over B
    dim3 gAttn(S / 256, S / 128, B);    // (4, 8, 8)
    gemm_wmma_bf16<0, true, 1024><<<gAttn, 256, 0, stream>>>(Qb, Kb, Sc, D, D, S, SD, SD, SS, 0.03125f);

    // 5) row softmax -> bf16 probabilities
    softmax_rows<<<B * S, 256, 0, stream>>>(Sc, P, S);

    // 6) AO = P V  (B operand = V^T rows, ldb = B*S, batch offset = b*S columns)
    gemm_wmma_bf16<1, false, 1024><<<gAttn, 256, 0, stream>>>(P, VT, AO, S, /*ldb=*/BS, D, SS, /*sB=*/S, SD, 1.0f);

    // 7) out = AO Wo^T  (f32 into d_out)
    gemm_wmma_bf16<0, false, 1024><<<gProj, 256, 0, stream>>>(AO, Wob, out, D, D, D, 0, 0, 0, 1.0f);
}
